// Actor_71777493451338
// MI455X (gfx1250) — matrix operand order repaired and checked by reading, not compiled, not run on hardware
//
#include <hip/hip_runtime.h>
#include <hip/hip_bf16.h>

typedef __attribute__((ext_vector_type(16))) _Float16 v16h;
typedef __attribute__((ext_vector_type(8)))  _Float16 v8h;
typedef __attribute__((ext_vector_type(8)))  float    v8f;

#define T_   256
#define B_   256
#define OBS_ 32
#define H_   512
#define G_   2048      // 4*H
#define M1_  512
#define M2_  256
#define A_   8
#define TB_  65536     // T*B
#define LOG2PI_ 1.8378770664093453f

__device__ __forceinline__ float sigm(float x){ return 1.0f / (1.0f + __expf(-x)); }
__device__ __forceinline__ float eluf(float x){ return x > 0.0f ? x : (__expf(x) - 1.0f); }

// ---------- WMMA fragment loaders (wave32, 16x16x32 f16) ----------
// A fragment: 16x32 (MxK), source row-major [M][K] f16.
// Lane layout: lanes 0-15 hold row M=lane, K=0..15; lanes 16-31 same row, K=16..31 (matches load_b).
__device__ __forceinline__ v16h load_a(const _Float16* p, int ld, int m0, int k0, int lane){
  const int row = m0 + (lane & 15);
  const int off = (lane >> 4) * 16;
  const _Float16* base = p + (size_t)row * ld + k0 + off;
  v8h lo = *(const v8h*)(base);
  v8h hi = *(const v8h*)(base + 8);
  v16h r;
#pragma unroll
  for (int i = 0; i < 8; ++i) { r[i] = lo[i]; r[i + 8] = hi[i]; }
  return r;
}

// B fragment: 32x16 (KxN) where B = W^T and W is row-major [N][K] f16.
// Lane layout: N = lane&15; lanes 0-15 K=0..15, lanes 16-31 K=16..31 -> one 32B contiguous load.
__device__ __forceinline__ v16h load_b(const _Float16* W, int ldw, int n0, int k0, int lane){
  const int col = n0 + (lane & 15);
  const _Float16* base = W + (size_t)col * ldw + k0 + (lane >> 4) * 16;
  return *(const v16h*)base;
}

// ---------- prep kernels ----------
__global__ void k_f32_to_f16(const float* __restrict__ src, _Float16* __restrict__ dst, int n){
  int i = blockIdx.x * blockDim.x + threadIdx.x;
  if (i < n) dst[i] = (_Float16)src[i];
}

__global__ void k_bias_sum(const float* __restrict__ bi, const float* __restrict__ bh,
                           float* __restrict__ out, int n){
  int i = blockIdx.x * blockDim.x + threadIdx.x;
  if (i < n) out[i] = bi[i] + bh[i];
}

// pack [Wm(8x256); Ws(8x256)] into one 16x256 f16 B-source
__global__ void k_head_pack(const float* __restrict__ Wm, const float* __restrict__ Ws,
                            _Float16* __restrict__ dst){
  int i = blockIdx.x * blockDim.x + threadIdx.x;   // 16*256
  if (i >= 16 * M2_) return;
  int r = i >> 8, k = i & 255;
  float v = (r < 8) ? Wm[r * M2_ + k] : Ws[(r - 8) * M2_ + k];
  dst[i] = (_Float16)v;
}

// h16 is stored PRE-MASKED with done[0] so k_gates needs no masking on its critical path
__global__ void k_copy_state(const float* __restrict__ h0, const float* __restrict__ c0,
                             const float* __restrict__ done,
                             float* __restrict__ h, float* __restrict__ c,
                             _Float16* __restrict__ h16){
  int i = blockIdx.x * blockDim.x + threadIdx.x;   // B*H
  if (i >= B_ * H_) return;
  int b = i >> 9;
  float hv = h0[i];
  float m0 = 1.0f - done[b];                        // done[t=0]
  h[i] = hv; h16[i] = (_Float16)(hv * m0); c[i] = c0[i];
}

// ---------- LSTM step: gates GEMM  gates = h_masked@Whh^T + x_t@Wih^T + (b_ih+b_hh) ----------
// h16 arrives already masked for this step -> inner loop is pure load+wmma streaming.
__global__ __launch_bounds__(128) void k_gates(
    const _Float16* __restrict__ h16,   // [B][H] pre-masked
    const _Float16* __restrict__ x16,   // [T*B][OBS]
    const _Float16* __restrict__ Whh,   // [4H][H]
    const _Float16* __restrict__ Wih,   // [4H][OBS]
    const float*    __restrict__ biasg, // [4H]
    float*          __restrict__ gates, // [B][4H]
    int t)
{
  const int lane = threadIdx.x & 31;
  const int wave = threadIdx.x >> 5;
  const int n0 = blockIdx.x * 64 + wave * 16;   // column tile (gate index)
  const int m0 = blockIdx.y * 16;               // row tile (batch index)
  const int col = n0 + (lane & 15);

  v8f acc = {};
  // recurrent contribution, K = 512
  for (int k0 = 0; k0 < H_; k0 += 32) {
    v16h a = load_a(h16, H_, m0, k0, lane);
    v16h b = load_b(Whh, H_, n0, k0, lane);
    __builtin_prefetch(Whh + (size_t)col * H_ + k0 + 64, 0, 1);
    acc = __builtin_amdgcn_wmma_f32_16x16x32_f16(false, a, false, b, (short)0, acc, false, false);
  }
  // input contribution, K = 32 (single WMMA)
  {
    v16h a = load_a(x16 + (size_t)t * B_ * OBS_, OBS_, m0, 0, lane);
    v16h b = load_b(Wih, OBS_, n0, 0, lane);
    acc = __builtin_amdgcn_wmma_f32_16x16x32_f16(false, a, false, b, (short)0, acc, false, false);
  }

  const float bias = biasg[col];
  const int rbase = m0 + (lane >> 4) * 8;
#pragma unroll
  for (int r = 0; r < 8; ++r)
    gates[(size_t)(rbase + r) * G_ + col] = acc[r] + bias;
}

// ---------- LSTM step: elementwise cell update ----------
// Writes h16 pre-masked with done[t+1] (mask for the NEXT step's gates GEMM);
// f32 h/c/hs keep unmasked values for hs[t] and the final hT/cT outputs.
__global__ __launch_bounds__(512) void k_update(
    const float* __restrict__ gates, const float* __restrict__ done,
    float* __restrict__ h, float* __restrict__ c,
    _Float16* __restrict__ h16, float* __restrict__ hs, int t)
{
  int idx = blockIdx.x * blockDim.x + threadIdx.x; // B*H exactly
  int b = idx >> 9, j = idx & 511;
  float m  = 1.0f - done[t * B_ + b];
  const float* g = gates + (size_t)b * G_;
  float ig = g[j], fg = g[H_ + j], gg = g[2 * H_ + j], og = g[3 * H_ + j];
  float cp = c[idx] * m;
  float cn = sigm(fg) * cp + sigm(ig) * tanhf(gg);
  float hn = sigm(og) * tanhf(cn);
  float mn = (t < T_ - 1) ? (1.0f - done[(t + 1) * B_ + b]) : 1.0f;
  c[idx] = cn; h[idx] = hn; h16[idx] = (_Float16)(hn * mn);
  hs[(size_t)t * (B_ * H_) + idx] = hn;
}

// ---------- fused LN + MLP + heads + logp/entropy, 16 rows per workgroup ----------
__global__ __launch_bounds__(256) void k_mlp(
    const float* __restrict__ hs,       // [TB][H]
    const float* __restrict__ g_ln, const float* __restrict__ b_ln,
    const _Float16* __restrict__ W1_16, const float* __restrict__ b1,
    const _Float16* __restrict__ W2_16, const float* __restrict__ b2,
    const _Float16* __restrict__ headW, // [16][256] packed Wm;Ws
    const float* __restrict__ bm, const float* __restrict__ bs,
    const float* __restrict__ action,   // [TB][A]
    float* __restrict__ out_logp, float* __restrict__ out_ent,
    float* __restrict__ out_mean, float* __restrict__ out_std)
{
  __shared__ __align__(32) _Float16 s_ln[16 * 520];
  __shared__ __align__(32) _Float16 s_h1[16 * 520];
  __shared__ __align__(32) _Float16 s_h2[16 * 264];
  __shared__ float s_msl[16 * 16];

  const int lane = threadIdx.x & 31;
  const int wave = threadIdx.x >> 5;           // 8 waves
  const size_t row0 = (size_t)blockIdx.x * 16;

  // ---- LayerNorm: wave w handles rows 2w, 2w+1 ----
#pragma unroll
  for (int rr = 0; rr < 2; ++rr) {
    int r = wave * 2 + rr;
    const float* src = hs + (row0 + r) * H_ + lane * 16;
    float v[16]; float s = 0.f, s2 = 0.f;
#pragma unroll
    for (int i = 0; i < 16; ++i) { float x = src[i]; v[i] = x; s += x; s2 += x * x; }
#pragma unroll
    for (int off = 16; off; off >>= 1) { s += __shfl_xor(s, off, 32); s2 += __shfl_xor(s2, off, 32); }
    float mu = s * (1.0f / H_);
    float var = s2 * (1.0f / H_) - mu * mu;
    float rstd = rsqrtf(var + 1e-5f);
#pragma unroll
    for (int i = 0; i < 16; ++i) {
      int j = lane * 16 + i;
      s_ln[r * 520 + j] = (_Float16)((v[i] - mu) * rstd * g_ln[j] + b_ln[j]);
    }
  }
  __syncthreads();

  // ---- GEMM1: [16x512] x [512x512] + b1, ELU -> s_h1 ; wave handles 64 cols ----
  {
    v8f acc[4] = {};
    const int colb = wave * 64;
    for (int k0 = 0; k0 < H_; k0 += 32) {
      v16h a = load_a(s_ln, 520, 0, k0, lane);
#pragma unroll
      for (int nt = 0; nt < 4; ++nt) {
        v16h b = load_b(W1_16, H_, colb + nt * 16, k0, lane);
        acc[nt] = __builtin_amdgcn_wmma_f32_16x16x32_f16(false, a, false, b, (short)0, acc[nt], false, false);
      }
    }
    const int rb = (lane >> 4) * 8;
#pragma unroll
    for (int nt = 0; nt < 4; ++nt) {
      int col = colb + nt * 16 + (lane & 15);
      float bias = b1[col];
#pragma unroll
      for (int r = 0; r < 8; ++r)
        s_h1[(rb + r) * 520 + col] = (_Float16)eluf(acc[nt][r] + bias);
    }
  }
  __syncthreads();

  // ---- GEMM2: [16x512] x [512x256] + b2, ELU -> s_h2 ; wave handles 32 cols ----
  {
    v8f acc[2] = {};
    const int colb = wave * 32;
    for (int k0 = 0; k0 < M1_; k0 += 32) {
      v16h a = load_a(s_h1, 520, 0, k0, lane);
#pragma unroll
      for (int nt = 0; nt < 2; ++nt) {
        v16h b = load_b(W2_16, M1_, colb + nt * 16, k0, lane);
        acc[nt] = __builtin_amdgcn_wmma_f32_16x16x32_f16(false, a, false, b, (short)0, acc[nt], false, false);
      }
    }
    const int rb = (lane >> 4) * 8;
#pragma unroll
    for (int nt = 0; nt < 2; ++nt) {
      int col = colb + nt * 16 + (lane & 15);
      float bias = b2[col];
#pragma unroll
      for (int r = 0; r < 8; ++r)
        s_h2[(rb + r) * 264 + col] = (_Float16)eluf(acc[nt][r] + bias);
    }
  }
  __syncthreads();

  // ---- GEMM3 (heads): [16x256] x [256x16] -> cols 0-7 mean-pre, 8-15 logstd-pre ----
  if (wave == 0) {
    v8f acc = {};
    for (int k0 = 0; k0 < M2_; k0 += 32) {
      v16h a = load_a(s_h2, 264, 0, k0, lane);
      v16h b = load_b(headW, M2_, 0, k0, lane);
      acc = __builtin_amdgcn_wmma_f32_16x16x32_f16(false, a, false, b, (short)0, acc, false, false);
    }
    const int rb = (lane >> 4) * 8, col = lane & 15;
#pragma unroll
    for (int r = 0; r < 8; ++r)
      s_msl[(rb + r) * 16 + col] = acc[r];
  }
  __syncthreads();

  // ---- per-row epilogue: heads bias, clip, exp, logp, entropy ----
  if (threadIdx.x < 16) {
    int r = threadIdx.x;
    size_t row = row0 + r;
    const float* act = action + row * A_;
    float logp = 0.f, entl = 0.f;
#pragma unroll
    for (int a = 0; a < A_; ++a) {
      float mean = s_msl[r * 16 + a] + bm[a];
      float ls = s_msl[r * 16 + 8 + a] + bs[a];
      ls = fminf(fmaxf(ls, -5.0f), 2.0f);
      float sd = __expf(ls);
      out_mean[row * A_ + a] = mean;
      out_std[row * A_ + a]  = sd;
      float z = (act[a] - mean) / sd;
      logp += -0.5f * z * z - ls - 0.5f * LOG2PI_;
      entl += ls;
    }
    out_logp[row] = logp;
    out_ent[row]  = entl + (float)A_ * (0.5f + 0.5f * LOG2PI_);
  }
}

// ======================= host launcher =======================
extern "C" void kernel_launch(void* const* d_in, const int* in_sizes, int n_in,
                              void* d_out, int out_size, void* d_ws, size_t ws_size,
                              hipStream_t stream) {
  const float* x      = (const float*)d_in[0];
  const float* done   = (const float*)d_in[1];
  const float* h0     = (const float*)d_in[2];
  const float* c0     = (const float*)d_in[3];
  const float* action = (const float*)d_in[4];
  const float* W_ih   = (const float*)d_in[5];
  const float* W_hh   = (const float*)d_in[6];
  const float* b_ih   = (const float*)d_in[7];
  const float* b_hh   = (const float*)d_in[8];
  const float* g_ln   = (const float*)d_in[9];
  const float* b_ln   = (const float*)d_in[10];
  const float* W1     = (const float*)d_in[11];
  const float* b1     = (const float*)d_in[12];
  const float* W2     = (const float*)d_in[13];
  const float* b2     = (const float*)d_in[14];
  const float* Wm     = (const float*)d_in[15];
  const float* bm     = (const float*)d_in[16];
  const float* Ws     = (const float*)d_in[17];
  const float* bs     = (const float*)d_in[18];

  // workspace layout (bytes, all 256-aligned)
  char* base = (char*)d_ws;
  float*     hs     = (float*)(base + 0);                         // 134217728
  float*     gates  = (float*)(base + 134217728);                 // 2097152
  float*     hbuf   = (float*)(base + 136314880);                 // 524288
  float*     cbuf   = (float*)(base + 136839168);                 // 524288
  _Float16*  h16    = (_Float16*)(base + 137363456);              // 262144
  _Float16*  x16    = (_Float16*)(base + 137625600);              // 4194304
  _Float16*  Wih16  = (_Float16*)(base + 141819904);              // 131072
  _Float16*  Whh16  = (_Float16*)(base + 141950976);              // 2097152
  _Float16*  W1_16  = (_Float16*)(base + 144048128);              // 524288
  _Float16*  W2_16  = (_Float16*)(base + 144572416);              // 262144
  _Float16*  headW  = (_Float16*)(base + 144834560);              // 8192
  float*     biasg  = (float*)(base + 144842752);                 // 8192

  // output layout (floats): action | logp | ent | hT | cT | mean | std
  float* out_action = (float*)d_out;
  float* out_logp   = out_action + (size_t)TB_ * A_;    // 524288
  float* out_ent    = out_logp + TB_;                   // 589824
  float* out_hT     = out_ent + TB_;                    // 655360
  float* out_cT     = out_hT + B_ * H_;                 // 786432
  float* out_mean   = out_cT + B_ * H_;                 // 917504
  float* out_std    = out_mean + (size_t)TB_ * A_;      // 1441792

  auto cvt = [&](const float* s, _Float16* d, int n) {
    k_f32_to_f16<<<(n + 255) / 256, 256, 0, stream>>>(s, d, n);
  };
  cvt(x,    x16,   TB_ * OBS_);
  cvt(W_ih, Wih16, G_ * OBS_);
  cvt(W_hh, Whh16, G_ * H_);
  cvt(W1,   W1_16, M1_ * H_);
  cvt(W2,   W2_16, M2_ * M1_);
  k_bias_sum<<<(G_ + 255) / 256, 256, 0, stream>>>(b_ih, b_hh, biasg, G_);
  k_head_pack<<<(16 * M2_ + 255) / 256, 256, 0, stream>>>(Wm, Ws, headW);
  k_copy_state<<<(B_ * H_ + 255) / 256, 256, 0, stream>>>(h0, c0, done, hbuf, cbuf, h16);

  // sequential LSTM scan: 2 kernels per timestep
  dim3 ggrid(G_ / 64, B_ / 16);   // 32 x 16
  for (int t = 0; t < T_; ++t) {
    k_gates<<<ggrid, 128, 0, stream>>>(h16, x16, Whh16, Wih16, biasg, gates, t);
    k_update<<<B_, 512, 0, stream>>>(gates, done, hbuf, cbuf, h16, hs, t);
  }

  // fused LN + MLP + heads + logp/ent over all T*B rows
  k_mlp<<<TB_ / 16, 256, 0, stream>>>(hs, g_ln, b_ln, W1_16, b1, W2_16, b2,
                                      headW, bm, bs, action,
                                      out_logp, out_ent, out_mean, out_std);

  // passthrough outputs
  hipMemcpyAsync(out_action, action, sizeof(float) * (size_t)TB_ * A_, hipMemcpyDeviceToDevice, stream);
  hipMemcpyAsync(out_hT, hbuf, sizeof(float) * (size_t)B_ * H_, hipMemcpyDeviceToDevice, stream);
  hipMemcpyAsync(out_cT, cbuf, sizeof(float) * (size_t)B_ * H_, hipMemcpyDeviceToDevice, stream);
}
